// SpikingPolicyNetwork_90108413870275
// MI455X (gfx1250) — compile-verified, matrix operand order checked by reading, not run on hardware
//
#include <hip/hip_runtime.h>
#include <hip/hip_bf16.h>
#include <math.h>

#define B_ 1024
#define S_ 512
#define H_ 1024
#define T_ 32

typedef __attribute__((ext_vector_type(16))) _Float16 v16h;
typedef __attribute__((ext_vector_type(8)))  _Float16 v8h;
typedef __attribute__((ext_vector_type(8)))  float    v8f;

union V16 { v16h v; v8h h[2]; };

__device__ __forceinline__ float sigm(float x) { return 1.0f / (1.0f + expf(-x)); }

// ---------------------------------------------------------------------------
// Encoder: state (B,S) f32 -> spikes0 (T,B,S) fp16.  One thread per (b,s),
// sequential scan over T; writes are contiguous per-t (coalesced).
// ---------------------------------------------------------------------------
__global__ __launch_bounds__(256)
void encode_kernel(const float* __restrict__ state,
                   const float* __restrict__ beta_raw,
                   const float* __restrict__ ethr,
                   _Float16* __restrict__ spk0)
{
    int idx = blockIdx.x * 256 + threadIdx.x;      // b*S + s
    if (idx >= B_ * S_) return;
    int s = idx & (S_ - 1);
    float z    = state[idx];
    float beta = sigm(beta_raw[s]);
    float th   = fabsf(ethr[s]);
    float base = beta * z;
    float amp  = (1.0f - beta) * z;
    float w    = (2.0f * 3.14159265358979323846f / (float)T_) * (float)s;
    float m = 0.0f;
    for (int t = 0; t < T_; ++t) {
        float c = base + amp * sinf(w * (float)t);
        m += c;
        float sp = (m - th >= 0.0f) ? 1.0f : 0.0f;
        m *= (1.0f - sp);
        spk0[(size_t)t * (B_ * S_) + idx] = (_Float16)sp;
    }
}

// ---------------------------------------------------------------------------
// Fused LIF layer, 2x-M register tiling:
//   block tile = 32 b x 128 h, 256 threads = 8 waves, each wave owns 16 cols
//   and TWO 16x16 accumulators (M rows 0-15 / 16-31)  -> one B fragment is
//   reused by two WMMAs: 6 ds_load_b128 per 2 WMMAs instead of 4 per 1.
//   W tile (128 x K) lives in LDS across all T steps; A tile (32 x K spikes)
//   re-staged per t.  LDS = (32+128)*K halfs = 320KB at K=1024 (full WGP LDS).
//   Membrane scan fused in C/D-layout registers; spikes emitted as fp16.
// ---------------------------------------------------------------------------
template<int K>
__global__ __launch_bounds__(256)
void lif_layer_kernel(const _Float16* __restrict__ spk_in,   // (T,B,K)
                      const float*    __restrict__ W,        // (H,K) row-major
                      const float*    __restrict__ decay_raw,// (H)
                      const float*    __restrict__ thr,      // (H)
                      _Float16*       __restrict__ spk_out)  // (T,B,H)
{
    extern __shared__ _Float16 smem[];
    _Float16* Ash = smem;             // 32*K
    _Float16* Bsh = smem + 32 * K;    // 128*K

    const int b0  = blockIdx.x * 32;
    const int h0  = blockIdx.y * 128;
    const int tid = threadIdx.x;      // 0..255

    // ---- load W tile (128 x K) f32 -> f16 into LDS, coalesced over k ----
    for (int i = tid; i < 128 * K; i += 256) {
        int n = i / K, k = i - n * K;
        Bsh[i] = (_Float16)W[(size_t)(h0 + n) * K + k];
    }

    const int lane = tid & 31;
    const int wave = tid >> 5;
    const int rrow = lane & 15;                 // A row / B col / D col (n)
    const int k0   = (lane & 16) ? 8 : 0;       // half-wave K offset
    const int ncol = h0 + wave * 16 + rrow;     // global output column

    const float dec = sigm(decay_raw[ncol]);
    const float th  = fabsf(thr[ncol]);

    v8f mstate0 = {};                           // membrane, M rows 0-15
    v8f mstate1 = {};                           // membrane, M rows 16-31

    const _Float16* Abase0 = Ash + rrow * K + k0;            // M tile 0
    const _Float16* Abase1 = Ash + (16 + rrow) * K + k0;     // M tile 1
    const _Float16* Bbase  = Bsh + (wave * 16 + rrow) * K + k0;
    const int mbase = b0 + ((lane & 16) ? 8 : 0);

    __syncthreads();

    for (int t = 0; t < T_; ++t) {
        // stage A tile: rows b0..b0+31 at time t are a contiguous 32*K block
        const float4* src = (const float4*)(spk_in + ((size_t)t * B_ + b0) * K);
        float4* dstA = (float4*)Ash;
        for (int i = tid; i < (32 * K) / 8; i += 256) dstA[i] = src[i];
        __syncthreads();

        // prefetch next t's A tile into cache while we do the WMMA loop
        if (t + 1 < T_) {
            const char* nxt = (const char*)(spk_in + ((size_t)(t + 1) * B_ + b0) * K);
            __builtin_prefetch(nxt + tid * ((32 * K * 2) / 256), 0, 0);
        }

        v8f acc0 = {};
        v8f acc1 = {};
        #pragma unroll 4
        for (int kk = 0; kk < K / 32; ++kk) {
            V16 a0, a1, b;
            b.h[0]  = *(const v8h*)(Bbase  + kk * 32);
            b.h[1]  = *(const v8h*)(Bbase  + kk * 32 + 16);
            a0.h[0] = *(const v8h*)(Abase0 + kk * 32);
            a0.h[1] = *(const v8h*)(Abase0 + kk * 32 + 16);
            a1.h[0] = *(const v8h*)(Abase1 + kk * 32);
            a1.h[1] = *(const v8h*)(Abase1 + kk * 32 + 16);
            acc0 = __builtin_amdgcn_wmma_f32_16x16x32_f16(
                     false, a0.v, false, b.v, (short)0, acc0, false, false);
            acc1 = __builtin_amdgcn_wmma_f32_16x16x32_f16(
                     false, a1.v, false, b.v, (short)0, acc1, false, false);
        }
        __syncthreads();   // A tile consumed; safe to overwrite next t

        // membrane scan + spike emission (C layout: vgpr r -> m-row)
        _Float16* dst = spk_out + (size_t)t * B_ * H_;
        #pragma unroll
        for (int r = 0; r < 8; ++r) {
            float mv = dec * mstate0[r] + acc0[r];
            float sp = (mv - th >= 0.0f) ? 1.0f : 0.0f;
            mstate0[r] = mv * (1.0f - sp);
            dst[(size_t)(mbase + r) * H_ + ncol] = (_Float16)sp;
        }
        #pragma unroll
        for (int r = 0; r < 8; ++r) {
            float mv = dec * mstate1[r] + acc1[r];
            float sp = (mv - th >= 0.0f) ? 1.0f : 0.0f;
            mstate1[r] = mv * (1.0f - sp);
            dst[(size_t)(mbase + 16 + r) * H_ + ncol] = (_Float16)sp;
        }
    }
}

// ---------------------------------------------------------------------------
// Heads: per-batch block. 4 recency-weighted aggregates over (T,H), then
// 33 GEMV rows (wave-per-row, shuffle reduce), then softmax/sigmoid/exp/clip.
// ---------------------------------------------------------------------------
__global__ __launch_bounds__(256)
void heads_kernel(const _Float16* __restrict__ spk,   // (T,B,H)
                  const float* offW, const float* offb, const float* offr,
                  const float* rmW,  const float* rmb,
                  const float* rsW,  const float* rsb, const float* rr,
                  const float* cW,   const float* cb,  const float* cr,
                  const float* fmW,  const float* fmb,
                  const float* fsW,  const float* fsb, const float* fr,
                  float* __restrict__ out)
{
    __shared__ float kap[4][T_];
    __shared__ float agg[4][H_];
    __shared__ float raw[33];

    const int b = blockIdx.x, tid = threadIdx.x;

    if (tid < 128) {
        int head = tid >> 5, t = tid & 31;
        const float* rp = (head == 0) ? offr : (head == 1) ? rr : (head == 2) ? cr : fr;
        kap[head][t] = expf(-fabsf(rp[0]) * (float)(T_ - 1 - t));
    }
    __syncthreads();
    if (tid < 4) {
        float s = 0.0f;
        for (int t = 0; t < T_; ++t) s += kap[tid][t];
        float inv = 1.0f / s;
        for (int t = 0; t < T_; ++t) kap[tid][t] *= inv;
    }
    __syncthreads();

    for (int h = tid; h < H_; h += 256) {
        float a0 = 0, a1 = 0, a2 = 0, a3 = 0;
        for (int t = 0; t < T_; ++t) {
            float v = (float)spk[((size_t)t * B_ + b) * H_ + h];
            a0 += v * kap[0][t]; a1 += v * kap[1][t];
            a2 += v * kap[2][t]; a3 += v * kap[3][t];
        }
        agg[0][h] = a0; agg[1][h] = a1; agg[2][h] = a2; agg[3][h] = a3;
    }
    __syncthreads();

    const int wv = tid >> 5, ln = tid & 31;
    for (int j = wv; j < 33; j += 8) {
        const float *wrow, *bias; int bi, ai;
        if (j < 17)      { wrow = offW + j * H_;        bias = offb; bi = j;      ai = 0; }
        else if (j < 19) { wrow = rmW + (j - 17) * H_;  bias = rmb;  bi = j - 17; ai = 1; }
        else if (j < 21) { wrow = rsW + (j - 19) * H_;  bias = rsb;  bi = j - 19; ai = 1; }
        else if (j < 29) { wrow = cW  + (j - 21) * H_;  bias = cb;   bi = j - 21; ai = 2; }
        else if (j < 31) { wrow = fmW + (j - 29) * H_;  bias = fmb;  bi = j - 29; ai = 3; }
        else             { wrow = fsW + (j - 31) * H_;  bias = fsb;  bi = j - 31; ai = 3; }
        float p = 0.0f;
        for (int h = ln; h < H_; h += 32) p += agg[ai][h] * wrow[h];
        for (int off = 16; off > 0; off >>= 1) p += __shfl_xor(p, off, 32);
        if (ln == 0) raw[j] = p + bias[bi];
    }
    __syncthreads();

    if (tid == 0) {
        float* o = out + (size_t)b * 58;
        // off head: logits + softmax (17)
        float mx = raw[0];
        for (int j = 1; j < 17; ++j) mx = fmaxf(mx, raw[j]);
        float e[17], s = 0.0f;
        for (int j = 0; j < 17; ++j) { e[j] = expf(raw[j] - mx); s += e[j]; }
        float inv = 1.0f / s;
        for (int j = 0; j < 17; ++j) { o[j] = raw[j]; o[17 + j] = e[j] * inv; }
        // res head
        o[34] = sigm(raw[17]); o[35] = sigm(raw[18]);
        o[36] = expf(fminf(fmaxf(raw[19], -5.0f), 2.0f));
        o[37] = expf(fminf(fmaxf(raw[20], -5.0f), 2.0f));
        // ch head: logits + softmax (8)
        float mx2 = raw[21];
        for (int j = 1; j < 8; ++j) mx2 = fmaxf(mx2, raw[21 + j]);
        float e2[8], s2 = 0.0f;
        for (int j = 0; j < 8; ++j) { e2[j] = expf(raw[21 + j] - mx2); s2 += e2[j]; }
        float inv2 = 1.0f / s2;
        for (int j = 0; j < 8; ++j) { o[38 + j] = raw[21 + j]; o[46 + j] = e2[j] * inv2; }
        // fl head
        o[54] = raw[29]; o[55] = raw[30];
        o[56] = expf(fminf(fmaxf(raw[31], -5.0f), 2.0f));
        o[57] = expf(fminf(fmaxf(raw[32], -5.0f), 2.0f));
    }
}

// ---------------------------------------------------------------------------
extern "C" void kernel_launch(void* const* d_in, const int* in_sizes, int n_in,
                              void* d_out, int out_size, void* d_ws, size_t ws_size,
                              hipStream_t stream)
{
    (void)n_in; (void)out_size; (void)ws_size;
    const float* const* f = (const float* const*)d_in;

    int i_state, i_beta, i_eth, i_w0, i_d0, i_t0, i_w1, i_d1, i_t1, i_w2, i_d2, i_t2,
        i_offW, i_offb, i_offr, i_rmW, i_rmb, i_rsW, i_rsb, i_rr,
        i_cW, i_cb, i_cr, i_fmW, i_fmb, i_fsW, i_fsb, i_fr;

    if (in_sizes[0] == B_ * S_) {
        // dict insertion order: state first
        i_state = 0; i_beta = 1; i_eth = 2;
        i_w0 = 3;  i_d0 = 4;  i_t0 = 5;
        i_w1 = 6;  i_d1 = 7;  i_t1 = 8;
        i_w2 = 9;  i_d2 = 10; i_t2 = 11;
        i_offr = 12; i_offW = 13; i_offb = 14;
        i_rr = 15; i_rmW = 16; i_rmb = 17; i_rsW = 18; i_rsb = 19;
        i_cr = 20; i_cW = 21; i_cb = 22;
        i_fr = 23; i_fmW = 24; i_fmb = 25; i_fsW = 26; i_fsb = 27;
    } else {
        // jax tree-flatten (sorted dict keys): 'params' < 'state'
        i_cW = 0; i_cb = 1; i_cr = 2; i_beta = 3; i_eth = 4;
        i_fmW = 5; i_fmb = 6; i_fr = 7; i_fsW = 8; i_fsb = 9;
        i_w0 = 10; i_d0 = 11; i_t0 = 12;
        i_w1 = 13; i_d1 = 14; i_t1 = 15;
        i_w2 = 16; i_d2 = 17; i_t2 = 18;
        i_offW = 19; i_offb = 20; i_offr = 21;
        i_rmW = 22; i_rmb = 23; i_rr = 24; i_rsW = 25; i_rsb = 26;
        i_state = 27;
    }

    // workspace layout (fp16): spk0 (T,B,S) | spk1 (T,B,H) | spk2 (T,B,H)
    // layer-3 output reuses spk1's buffer.
    _Float16* spk0 = (_Float16*)d_ws;
    _Float16* spk1 = spk0 + (size_t)T_ * B_ * S_;
    _Float16* spk2 = spk1 + (size_t)T_ * B_ * H_;
    _Float16* spk3 = spk1;

    encode_kernel<<<(B_ * S_) / 256, 256, 0, stream>>>(
        f[i_state], f[i_beta], f[i_eth], spk0);

    dim3 grid(B_ / 32, H_ / 128);
    lif_layer_kernel<512><<<grid, 256, (size_t)(32 + 128) * 512 * sizeof(_Float16), stream>>>(
        spk0, f[i_w0], f[i_d0], f[i_t0], spk1);
    lif_layer_kernel<1024><<<grid, 256, (size_t)(32 + 128) * 1024 * sizeof(_Float16), stream>>>(
        spk1, f[i_w1], f[i_d1], f[i_t1], spk2);
    lif_layer_kernel<1024><<<grid, 256, (size_t)(32 + 128) * 1024 * sizeof(_Float16), stream>>>(
        spk2, f[i_w2], f[i_d2], f[i_t2], spk3);

    heads_kernel<<<B_, 256, 0, stream>>>(
        spk3,
        f[i_offW], f[i_offb], f[i_offr],
        f[i_rmW], f[i_rmb], f[i_rsW], f[i_rsb], f[i_rr],
        f[i_cW], f[i_cb], f[i_cr],
        f[i_fmW], f[i_fmb], f[i_fsW], f[i_fsb], f[i_fr],
        (float*)d_out);
}
